// RNAGraphTransformerWithRefinement_21311627722960
// MI455X (gfx1250) — matrix operand order repaired and checked by reading, not compiled, not run on hardware
//
#include <hip/hip_runtime.h>
#include <hip/hip_bf16.h>
#include <math.h>

// ---------------------------------------------------------------------------
// Types for CDNA5 WMMA (wave32)
// ---------------------------------------------------------------------------
typedef __bf16 bf16;
typedef __attribute__((ext_vector_type(16))) __bf16 bf16x16;
typedef __attribute__((ext_vector_type(8)))  __bf16 bf16x8;
typedef __attribute__((ext_vector_type(8)))  float  f32x8;

#define CDIV(a, b) (((a) + (b) - 1) / (b))

static const int NN = 2048;   // nodes
static const int EE = 65536;  // edges
static const int DD = 256;    // hidden
static const int HH = 8;      // heads

__device__ __forceinline__ float gelu_f(float x) {
  // tanh approximation (jax.nn.gelu default approximate=True)
  float x3 = x * x * x;
  return 0.5f * x * (1.0f + tanhf(0.7978845608028654f * (x + 0.044715f * x3)));
}

__device__ __forceinline__ bf16x16 pack16(bf16x8 a, bf16x8 b) {
  bf16x16 r;
#pragma unroll
  for (int i = 0; i < 8; i++) { r[i] = a[i]; r[i + 8] = b[i]; }
  return r;
}

__device__ __forceinline__ void atomicMaxF(float* addr, float v) {
  unsigned int* ai = (unsigned int*)addr;
  unsigned int old = *ai;
  while (v > __uint_as_float(old)) {
    unsigned int assumed = old;
    old = atomicCAS(ai, assumed, __float_as_uint(v));
    if (old == assumed) break;
  }
}

// ---------------------------------------------------------------------------
// WMMA GEMM: C[M,N] = act(A_bf16[M,K] @ WT_bf16[N,K]^T + bias[N]) (+resid)
// One wave computes a 32x64 tile: 2 M-tiles x 4 N-tiles = 8 accumulators,
// 8 back-to-back v_wmma_f32_16x16x32_bf16 per K-chunk, A reused 4x, B 2x.
// Requires: K % 32 == 0, M % 32 == 0, N % 64 == 0.
// A-operand layout (16-bit A 16x32): lane m=lane&15; segs [8h,+8),[16+8h,+8)
// B-operand layout: lane n=lane&15; segs [8h,+8),[16+8h,+8) of WT row (same as A).
// C/D layout: lane n=lane&15; element v -> row m0+v+8h.
// ---------------------------------------------------------------------------
template <int ACT>
__global__ __launch_bounds__(32) void k_gemm(
    const bf16* __restrict__ A, const bf16* __restrict__ WT,
    const float* __restrict__ bias, const float* __restrict__ resid,
    float* __restrict__ Cf, bf16* __restrict__ Cb, int M, int N, int K) {
  int lane = threadIdx.x & 31;
  int half = lane >> 4, idx = lane & 15;
  int m0 = blockIdx.y << 5, n0 = blockIdx.x << 6;
  const bf16* arow0 = A + (size_t)(m0 + idx) * K;
  const bf16* arow1 = A + (size_t)(m0 + 16 + idx) * K;
  const bf16* brow[4];
#pragma unroll
  for (int ni = 0; ni < 4; ni++)
    brow[ni] = WT + (size_t)(n0 + 16 * ni + idx) * K;

  f32x8 acc[2][4];
#pragma unroll
  for (int mi = 0; mi < 2; mi++)
#pragma unroll
    for (int ni = 0; ni < 4; ni++) acc[mi][ni] = (f32x8){};

  for (int k0 = 0; k0 < K; k0 += 32) {
    if (k0 + 32 < K) {  // gfx1250 global_prefetch_b8 for next A chunk
      __builtin_prefetch((const void*)(arow0 + k0 + 32), 0, 1);
      __builtin_prefetch((const void*)(arow1 + k0 + 32), 0, 1);
    }
    bf16x16 a0 = pack16(*(const bf16x8*)(arow0 + k0 + 8 * half),
                        *(const bf16x8*)(arow0 + k0 + 16 + 8 * half));
    bf16x16 a1 = pack16(*(const bf16x8*)(arow1 + k0 + 8 * half),
                        *(const bf16x8*)(arow1 + k0 + 16 + 8 * half));
#pragma unroll
    for (int ni = 0; ni < 4; ni++) {
      bf16x16 b = pack16(*(const bf16x8*)(brow[ni] + k0 + 8 * half),
                         *(const bf16x8*)(brow[ni] + k0 + 16 + 8 * half));
      acc[0][ni] = __builtin_amdgcn_wmma_f32_16x16x32_bf16(
          false, a0, false, b, (short)0, acc[0][ni], false, false);
      acc[1][ni] = __builtin_amdgcn_wmma_f32_16x16x32_bf16(
          false, a1, false, b, (short)0, acc[1][ni], false, false);
    }
  }
#pragma unroll
  for (int ni = 0; ni < 4; ni++) {
    float bv = bias ? bias[n0 + 16 * ni + idx] : 0.0f;
#pragma unroll
    for (int mi = 0; mi < 2; mi++) {
#pragma unroll
      for (int v = 0; v < 8; v++) {
        int m = m0 + 16 * mi + v + 8 * half;
        size_t off = (size_t)m * N + n0 + 16 * ni + idx;
        float val = acc[mi][ni][v] + bv;
        if (ACT == 1) val = gelu_f(val);
        if (resid) val += resid[off];
        if (Cf) Cf[off] = val;
        if (Cb) Cb[off] = (bf16)val;
      }
    }
  }
}

// ---------------------------------------------------------------------------
// Flash attention: one wave per (head, 16-query block). Per-head C = 32, so
// QK^T per 16x16 score tile is a single 16x16x32 bf16 WMMA. Online softmax,
// P transposed through LDS into A-operand layout, P@V via WMMA against
// pre-transposed VT[channel][key]. HASB: additive bias[q][k] (refinement).
// ---------------------------------------------------------------------------
template <int HASB>
__global__ __launch_bounds__(32) void k_attn(
    const bf16* __restrict__ Qb, const bf16* __restrict__ Kb,
    const bf16* __restrict__ VTb, const float* __restrict__ bias,
    float* __restrict__ O, int Nn) {
  __shared__ __align__(16) bf16 lds[16 * 32];
  int lane = threadIdx.x & 31;
  int half = lane >> 4, idx = lane & 15;
  int h = blockIdx.y;
  int q0 = blockIdx.x << 4;
  const float scale = 0.17677669529663687f;  // 1/sqrt(32)

  const bf16* qrow = Qb + (size_t)(q0 + idx) * DD + h * 32;
  bf16x16 qa = pack16(*(const bf16x8*)(qrow + 8 * half),
                      *(const bf16x8*)(qrow + 16 + 8 * half));
  f32x8 o0 = {}, o1 = {};
  float mrow[8], lrow[8];
#pragma unroll
  for (int v = 0; v < 8; v++) { mrow[v] = -3.0e38f; lrow[v] = 0.0f; }

  for (int k0 = 0; k0 < Nn; k0 += 32) {
    f32x8 s[2];
#pragma unroll
    for (int t = 0; t < 2; t++) {
      const bf16* krow = Kb + (size_t)(k0 + 16 * t + idx) * DD + h * 32 + 8 * half;
      bf16x16 kv = pack16(*(const bf16x8*)krow, *(const bf16x8*)(krow + 16));
      f32x8 z = {};
      z = __builtin_amdgcn_wmma_f32_16x16x32_bf16(false, qa, false, kv,
                                                  (short)0, z, false, false);
      if (HASB) {
#pragma unroll
        for (int v = 0; v < 8; v++)
          z[v] = z[v] * scale +
                 bias[(size_t)(q0 + v + 8 * half) * Nn + (k0 + 16 * t + idx)];
      } else {
#pragma unroll
        for (int v = 0; v < 8; v++) z[v] = z[v] * scale;
      }
      s[t] = z;
    }
    // online softmax: per-row reductions across the 16 lanes of each half
#pragma unroll
    for (int v = 0; v < 8; v++) {
      float mx = fmaxf(s[0][v], s[1][v]);
      for (int m = 8; m >= 1; m >>= 1) mx = fmaxf(mx, __shfl_xor(mx, m, 32));
      float nm = fmaxf(mrow[v], mx);
      float resc = __expf(mrow[v] - nm);
      mrow[v] = nm;
      float p0 = __expf(s[0][v] - nm);
      float p1 = __expf(s[1][v] - nm);
      s[0][v] = p0; s[1][v] = p1;
      float rs = p0 + p1;
      for (int m = 8; m >= 1; m >>= 1) rs += __shfl_xor(rs, m, 32);
      lrow[v] = lrow[v] * resc + rs;
      o0[v] *= resc; o1[v] *= resc;
    }
    // transpose P (C-layout) into A-operand layout via LDS
    __syncthreads();
#pragma unroll
    for (int v = 0; v < 8; v++) {
      int r = v + 8 * half;
      lds[r * 32 + idx] = (bf16)s[0][v];
      lds[r * 32 + 16 + idx] = (bf16)s[1][v];
    }
    __syncthreads();
    bf16x16 pa = pack16(*(const bf16x8*)(&lds[idx * 32 + 8 * half]),
                        *(const bf16x8*)(&lds[idx * 32 + 16 + 8 * half]));
    {
      const bf16* v0 = VTb + (size_t)(h * 32 + idx) * Nn + k0 + 8 * half;
      bf16x16 vb0 = pack16(*(const bf16x8*)v0, *(const bf16x8*)(v0 + 16));
      o0 = __builtin_amdgcn_wmma_f32_16x16x32_bf16(false, pa, false, vb0,
                                                   (short)0, o0, false, false);
      const bf16* v1 = VTb + (size_t)(h * 32 + 16 + idx) * Nn + k0 + 8 * half;
      bf16x16 vb1 = pack16(*(const bf16x8*)v1, *(const bf16x8*)(v1 + 16));
      o1 = __builtin_amdgcn_wmma_f32_16x16x32_bf16(false, pa, false, vb1,
                                                   (short)0, o1, false, false);
    }
  }
#pragma unroll
  for (int v = 0; v < 8; v++) {
    float inv = 1.0f / (lrow[v] + 1e-30f);
    size_t row = (size_t)(q0 + v + 8 * half) * DD + h * 32;
    O[row + idx] = o0[v] * inv;
    O[row + 16 + idx] = o1[v] * inv;
  }
}

// ---------------------------------------------------------------------------
// LayerNorm over D=256 cols: one wave per row. y = LN(x + pre)*g + b (+ post)
// ---------------------------------------------------------------------------
__global__ __launch_bounds__(256) void k_layernorm(
    const float* __restrict__ x, const float* __restrict__ pre,
    const float* __restrict__ post, const float* __restrict__ g,
    const float* __restrict__ b, float* __restrict__ outf,
    bf16* __restrict__ outb) {
  int row = blockIdx.x * 8 + (threadIdx.x >> 5);
  int lane = threadIdx.x & 31;
  const float* xr = x + (size_t)row * DD;
  float vals[8];
  float s = 0.0f;
#pragma unroll
  for (int j = 0; j < 8; j++) {
    int c = lane + j * 32;
    float v = xr[c];
    if (pre) v += pre[(size_t)row * DD + c];
    vals[j] = v; s += v;
  }
  for (int m = 16; m >= 1; m >>= 1) s += __shfl_xor(s, m, 32);
  float mean = s * (1.0f / DD);
  float vs = 0.0f;
#pragma unroll
  for (int j = 0; j < 8; j++) { float d = vals[j] - mean; vs += d * d; }
  for (int m = 16; m >= 1; m >>= 1) vs += __shfl_xor(vs, m, 32);
  float inv = rsqrtf(vs * (1.0f / DD) + 1e-5f);
#pragma unroll
  for (int j = 0; j < 8; j++) {
    int c = lane + j * 32;
    float v = (vals[j] - mean) * inv * g[c] + b[c];
    if (post) v += post[(size_t)row * DD + c];
    size_t off = (size_t)row * DD + c;
    if (outf) outf[off] = v;
    if (outb) outb[off] = (bf16)v;
  }
}

// ---------------------------------------------------------------------------
// Small utility kernels
// ---------------------------------------------------------------------------
__global__ void k_wt_t(const float* __restrict__ W, bf16* __restrict__ WT,
                       int K, int N, int KT) {
  int t = blockIdx.x * blockDim.x + threadIdx.x;
  if (t >= N * KT) return;
  int n = t / KT, k = t % KT;
  WT[t] = (k < K) ? (bf16)W[(size_t)k * N + n] : (bf16)0.0f;
}

__global__ void k_act_t(const float* __restrict__ X, bf16* __restrict__ XT,
                        int M, int N) {
  int t = blockIdx.x * blockDim.x + threadIdx.x;
  if (t >= M * N) return;
  int n = t / N, c = t % N;
  XT[(size_t)c * M + n] = (bf16)X[t];
}

__global__ void k_f2b(const float* __restrict__ x, bf16* __restrict__ y, int n) {
  int t = blockIdx.x * blockDim.x + threadIdx.x;
  if (t < n) y[t] = (bf16)x[t];
}

__global__ void k_fill(float* __restrict__ p, float v, int n) {
  int t = blockIdx.x * blockDim.x + threadIdx.x;
  if (t < n) p[t] = v;
}

__global__ void k_copy(const float* __restrict__ s, float* __restrict__ d, int n) {
  int t = blockIdx.x * blockDim.x + threadIdx.x;
  if (t < n) d[t] = s[t];
}

__global__ void k_pos_add(float* __restrict__ xb) {
  int t = blockIdx.x * blockDim.x + threadIdx.x;
  if (t >= NN * DD) return;
  int n = t / DD, d = t % DD;
  int i = d >> 1;
  float div = __expf((float)(2 * i) * (-9.210340371976184f / 256.0f));
  float a = (float)n * div;
  xb[t] += (d & 1) ? cosf(a) : sinf(a);
}

__global__ void k_add_gelu(const float* __restrict__ agg,
                           const float* __restrict__ bvec,
                           const float* __restrict__ resid,
                           float* __restrict__ out) {
  int t = blockIdx.x * blockDim.x + threadIdx.x;
  if (t >= NN * DD) return;
  int c = t % DD;
  out[t] = gelu_f(agg[t] + bvec[c] + resid[t]);
}

// GATv2 edge kernels -------------------------------------------------------
__global__ void k_gat_logit(const float* __restrict__ xl,
                            const float* __restrict__ xr,
                            const bf16* __restrict__ ee,
                            const int* __restrict__ src,
                            const int* __restrict__ dst,
                            const float* __restrict__ att,
                            float* __restrict__ logit,
                            float* __restrict__ mbuf) {
  int t = blockIdx.x * blockDim.x + threadIdx.x;
  if (t >= EE * HH) return;
  int e = t >> 3, h = t & 7;
  int s = src[e], d = dst[e];
  const float* xls = xl + (size_t)s * DD + h * 32;
  const float* xrd = xr + (size_t)d * DD + h * 32;
  const bf16* eer = ee + (size_t)e * DD + h * 32;
  const float* ah = att + h * 32;
  float acc = 0.0f;
#pragma unroll 8
  for (int c = 0; c < 32; c++) {
    float v = xls[c] + xrd[c] + (float)eer[c];
    v = v > 0.0f ? v : 0.2f * v;  // leaky_relu(0.2)
    acc += v * ah[c];
  }
  logit[t] = acc;
  atomicMaxF(&mbuf[d * HH + h], acc);
}

__global__ void k_gat_exp(const int* __restrict__ dst,
                          float* __restrict__ logit,
                          const float* __restrict__ mbuf,
                          float* __restrict__ den) {
  int t = blockIdx.x * blockDim.x + threadIdx.x;
  if (t >= EE * HH) return;
  int e = t >> 3, h = t & 7;
  int d = dst[e];
  float ex = __expf(logit[t] - mbuf[d * HH + h]);
  logit[t] = ex;
  atomicAdd(&den[d * HH + h], ex);
}

__global__ void k_gat_agg(const float* __restrict__ xl,
                          const int* __restrict__ src,
                          const int* __restrict__ dst,
                          const float* __restrict__ logit,
                          const float* __restrict__ den,
                          float* __restrict__ out) {
  int t = blockIdx.x * blockDim.x + threadIdx.x;
  if (t >= EE * HH) return;
  int e = t >> 3, h = t & 7;
  int s = src[e], d = dst[e];
  float alpha = logit[t] / (den[d * HH + h] + 1e-16f);
  const float* xls = xl + (size_t)s * DD + h * 32;
  float* op = out + (size_t)d * DD + h * 32;
#pragma unroll 8
  for (int c = 0; c < 32; c++) atomicAdd(&op[c], alpha * xls[c]);
}

// Refinement helpers -------------------------------------------------------
__global__ void k_struct_bias(const float* __restrict__ coords,
                              float* __restrict__ bias) {
  int t = blockIdx.x * blockDim.x + threadIdx.x;
  if (t >= NN * NN) return;
  int n = t / NN, m = t % NN;
  float dx = coords[n * 3 + 0] - coords[m * 3 + 0];
  float dy = coords[n * 3 + 1] - coords[m * 3 + 1];
  float dz = coords[n * 3 + 2] - coords[m * 3 + 2];
  float dist = sqrtf(dx * dx + dy * dy + dz * dz + 1e-8f);
  bias[t] = -fminf(dist, 20.0f) * (1.0f / 20.0f);
}

__global__ void k_concat_pad(const float* __restrict__ emb,
                             const float* __restrict__ coords,
                             bf16* __restrict__ hcat) {
  int t = blockIdx.x * blockDim.x + threadIdx.x;
  if (t >= NN * 288) return;
  int n = t / 288, c = t % 288;
  float v = (c < 256) ? emb[(size_t)n * DD + c]
          : (c < 259) ? coords[n * 3 + (c - 256)] : 0.0f;
  hcat[t] = (bf16)v;
}

__global__ void k_proj3(const float* __restrict__ X, const float* __restrict__ W,
                        const float* __restrict__ b, const float* __restrict__ base,
                        float* __restrict__ out, int Din) {
  int t = blockIdx.x * blockDim.x + threadIdx.x;
  if (t >= NN * 3) return;
  int n = t / 3, j = t % 3;
  float acc = b[j];
  const float* xr = X + (size_t)n * Din;
  for (int c = 0; c < Din; c++) acc += xr[c] * W[c * 3 + j];
  if (base) acc += base[t];
  out[t] = acc;
}

// ---------------------------------------------------------------------------
// Host orchestration
// ---------------------------------------------------------------------------
extern "C" void kernel_launch(void* const* d_in, const int* in_sizes, int n_in,
                              void* d_out, int out_size, void* d_ws, size_t ws_size,
                              hipStream_t stream) {
  (void)in_sizes; (void)n_in; (void)out_size; (void)ws_size;
  // ---- input map (recursive insertion order of setup_inputs dict) ----
  const float* x_in   = (const float*)d_in[0];   // [2048,64]
  const int*   ei     = (const int*)d_in[1];     // [2,65536]
  const float* eattr  = (const float*)d_in[2];   // [65536,32]
  const float* node_W = (const float*)d_in[3];
  const float* node_b = (const float*)d_in[4];
  const float* edge_W = (const float*)d_in[5];
  const float* edge_b = (const float*)d_in[6];
  auto gatp = [&](int l, int j) { return (const float*)d_in[7 + 7 * l + j]; };
  // gat j: 0=Wl 1=Wr 2=We 3=att 4=b 5=ln_g 6=ln_b
  const int tf0 = 7 + 7 * 8;  // 63
  auto tfp = [&](int t, int j) { return (const float*)d_in[tf0 + 16 * t + j]; };
  // tf j: 0=Wq 1=bq 2=Wk 3=bk 4=Wv 5=bv 6=Wo 7=bo 8=ln1g 9=ln1b 10=ln2g 11=ln2b
  //       12=W1 13=b1 14=W2 15=b2
  const int pe = tf0 + 32;  // 95
  const float* fln_g = (const float*)d_in[pe + 0];
  const float* fln_b = (const float*)d_in[pe + 1];
  const float* mlpW1 = (const float*)d_in[pe + 2];
  const float* mlpb1 = (const float*)d_in[pe + 3];
  const float* mlpW2 = (const float*)d_in[pe + 4];
  const float* mlpb2 = (const float*)d_in[pe + 5];
  const float* mlpW3 = (const float*)d_in[pe + 6];
  const float* mlpb3 = (const float*)d_in[pe + 7];
  const float* refW1 = (const float*)d_in[pe + 8];
  const float* refb1 = (const float*)d_in[pe + 9];
  const float* refW2 = (const float*)d_in[pe + 10];
  const float* refb2 = (const float*)d_in[pe + 11];
  const int* e_src = ei;
  const int* e_dst = ei + EE;

  // ---- workspace bump allocator ----
  char* wsp = (char*)d_ws;
  size_t off = 0;
  auto alloc = [&](size_t bytes) -> void* {
    off = (off + 255) & ~(size_t)255;
    void* p = wsp + off;
    off += bytes;
    return p;
  };

  // weights (bf16, transposed [N][K])
  bf16* nodeWT = (bf16*)alloc((size_t)DD * 64 * 2);
  bf16* edgeWT = (bf16*)alloc((size_t)DD * 32 * 2);
  bf16 *WlT[8], *WrT[8], *WeT[8];
  for (int l = 0; l < 8; l++) {
    WlT[l] = (bf16*)alloc((size_t)DD * DD * 2);
    WrT[l] = (bf16*)alloc((size_t)DD * DD * 2);
    WeT[l] = (bf16*)alloc((size_t)DD * DD * 2);
  }
  bf16 *WqT[2], *WkT[2], *WvT[2], *WoT[2], *W1T[2], *W2T[2];
  for (int t = 0; t < 2; t++) {
    WqT[t] = (bf16*)alloc((size_t)DD * DD * 2);
    WkT[t] = (bf16*)alloc((size_t)DD * DD * 2);
    WvT[t] = (bf16*)alloc((size_t)DD * DD * 2);
    WoT[t] = (bf16*)alloc((size_t)DD * DD * 2);
    W1T[t] = (bf16*)alloc((size_t)1024 * DD * 2);
    W2T[t] = (bf16*)alloc((size_t)DD * 1024 * 2);
  }
  bf16* mW1T = (bf16*)alloc((size_t)512 * DD * 2);
  bf16* mW2T = (bf16*)alloc((size_t)DD * 512 * 2);
  bf16* rW1T = (bf16*)alloc((size_t)DD * 288 * 2);

  // activations
  bf16* xbf    = (bf16*)alloc((size_t)NN * 64 * 2);
  bf16* eainb  = (bf16*)alloc((size_t)EE * 32 * 2);
  bf16* eabf   = (bf16*)alloc((size_t)EE * DD * 2);
  bf16* eebf   = (bf16*)alloc((size_t)EE * DD * 2);
  float* hh[9];
  for (int i = 0; i < 9; i++) hh[i] = (float*)alloc((size_t)NN * DD * 4);
  bf16*  hb    = (bf16*)alloc((size_t)NN * DD * 2);
  float* xl    = (float*)alloc((size_t)NN * DD * 4);
  float* xr    = (float*)alloc((size_t)NN * DD * 4);
  float* logit = (float*)alloc((size_t)EE * HH * 4);
  float* mbuf  = (float*)alloc((size_t)NN * HH * 4);
  float* den   = (float*)alloc((size_t)NN * HH * 4);
  float* agg   = (float*)alloc((size_t)NN * DD * 4);
  float* tmp   = (float*)alloc((size_t)NN * DD * 4);
  float* xb    = (float*)alloc((size_t)NN * DD * 4);
  float* xcur  = (float*)alloc((size_t)NN * DD * 4);
  bf16*  xcurb = (bf16*)alloc((size_t)NN * DD * 2);
  float* x2    = (float*)alloc((size_t)NN * DD * 4);
  bf16*  x2b   = (bf16*)alloc((size_t)NN * DD * 2);
  bf16*  qb    = (bf16*)alloc((size_t)NN * DD * 2);
  bf16*  kb    = (bf16*)alloc((size_t)NN * DD * 2);
  float* vf    = (float*)alloc((size_t)NN * DD * 4);
  bf16*  vtb   = (bf16*)alloc((size_t)NN * DD * 2);
  float* aof   = (float*)alloc((size_t)NN * DD * 4);
  bf16*  aob   = (bf16*)alloc((size_t)NN * DD * 2);
  float* obuf  = (float*)alloc((size_t)NN * DD * 4);
  bf16*  midb  = (bf16*)alloc((size_t)NN * 1024 * 2);
  float* fbuf  = (float*)alloc((size_t)NN * DD * 4);
  float* biasb = (float*)alloc((size_t)NN * NN * 4);
  bf16*  hcatb = (bf16*)alloc((size_t)NN * 288 * 2);
  bf16*  t1b   = (bf16*)alloc((size_t)NN * 512 * 2);
  float* t2f   = (float*)alloc((size_t)NN * DD * 4);
  float* d1f   = (float*)alloc((size_t)NN * DD * 4);
  float* c0    = (float*)alloc((size_t)NN * 3 * 4);
  float* c1    = (float*)alloc((size_t)NN * 3 * 4);

  // ---- helpers ----
  auto wt = [&](const float* W, bf16* WT, int K, int N, int KT) {
    k_wt_t<<<CDIV(N * KT, 256), 256, 0, stream>>>(W, WT, K, N, KT);
  };
  auto gemm = [&](const bf16* A, const bf16* WT, const float* bias,
                  const float* resid, float* Cf, bf16* Cb, int M, int N, int K,
                  int act) {
    dim3 g(N / 64, M / 32);
    if (act)
      k_gemm<1><<<g, 32, 0, stream>>>(A, WT, bias, resid, Cf, Cb, M, N, K);
    else
      k_gemm<0><<<g, 32, 0, stream>>>(A, WT, bias, resid, Cf, Cb, M, N, K);
  };
  auto ln = [&](const float* xv, const float* pre, const float* post,
                const float* g, const float* b, float* of, bf16* ob) {
    k_layernorm<<<NN / 8, 256, 0, stream>>>(xv, pre, post, g, b, of, ob);
  };

  // ---- weight prep ----
  wt(node_W, nodeWT, 64, DD, 64);
  wt(edge_W, edgeWT, 32, DD, 32);
  for (int l = 0; l < 8; l++) {
    wt(gatp(l, 0), WlT[l], DD, DD, DD);
    wt(gatp(l, 1), WrT[l], DD, DD, DD);
    wt(gatp(l, 2), WeT[l], DD, DD, DD);
  }
  for (int t = 0; t < 2; t++) {
    wt(tfp(t, 0), WqT[t], DD, DD, DD);
    wt(tfp(t, 2), WkT[t], DD, DD, DD);
    wt(tfp(t, 4), WvT[t], DD, DD, DD);
    wt(tfp(t, 6), WoT[t], DD, DD, DD);
    wt(tfp(t, 12), W1T[t], DD, 1024, DD);
    wt(tfp(t, 14), W2T[t], 1024, DD, 1024);
  }
  wt(mlpW1, mW1T, DD, 512, DD);
  wt(mlpW2, mW2T, 512, DD, 512);
  wt(refW1, rW1T, 259, DD, 288);  // pad K 259 -> 288 with zeros

  // ---- embeddings ----
  k_f2b<<<CDIV(NN * 64, 256), 256, 0, stream>>>(x_in, xbf, NN * 64);
  k_f2b<<<CDIV(EE * 32, 256), 256, 0, stream>>>(eattr, eainb, EE * 32);
  gemm(xbf, nodeWT, node_b, nullptr, hh[0], hb, NN, DD, 64, 0);
  gemm(eainb, edgeWT, edge_b, nullptr, nullptr, eabf, EE, DD, 32, 0);

  // ---- GATv2 stack ----
  for (int l = 0; l < 8; l++) {
    gemm(hb, WlT[l], nullptr, nullptr, xl, nullptr, NN, DD, DD, 0);
    gemm(hb, WrT[l], nullptr, nullptr, xr, nullptr, NN, DD, DD, 0);
    gemm(eabf, WeT[l], nullptr, nullptr, nullptr, eebf, EE, DD, DD, 0);
    k_fill<<<CDIV(NN * HH, 256), 256, 0, stream>>>(mbuf, -3.0e38f, NN * HH);
    k_fill<<<CDIV(NN * HH, 256), 256, 0, stream>>>(den, 0.0f, NN * HH);
    k_fill<<<CDIV(NN * DD, 256), 256, 0, stream>>>(agg, 0.0f, NN * DD);
    k_gat_logit<<<CDIV(EE * HH, 256), 256, 0, stream>>>(
        xl, xr, eebf, e_src, e_dst, gatp(l, 3), logit, mbuf);
    k_gat_exp<<<CDIV(EE * HH, 256), 256, 0, stream>>>(e_dst, logit, mbuf, den);
    k_gat_agg<<<CDIV(EE * HH, 256), 256, 0, stream>>>(xl, e_src, e_dst, logit,
                                                      den, agg);
    k_add_gelu<<<CDIV(NN * DD, 256), 256, 0, stream>>>(agg, gatp(l, 4), hh[l],
                                                       tmp);
    const float* post = (l >= 2 && (l % 2 == 0)) ? hh[l - 1] : nullptr;
    ln(tmp, nullptr, post, gatp(l, 5), gatp(l, 6), hh[l + 1], hb);
  }

  // ---- final LN + positional encoding -> xb (fixed encoder input) ----
  ln(hh[8], nullptr, nullptr, fln_g, fln_b, xb, nullptr);
  k_pos_add<<<CDIV(NN * DD, 256), 256, 0, stream>>>(xb);

  // ---- transformer encoder ----
  auto encoder = [&](const float* biasp) {
    hipMemcpyAsync(xcur, xb, (size_t)NN * DD * 4, hipMemcpyDeviceToDevice, stream);
    k_f2b<<<CDIV(NN * DD, 256), 256, 0, stream>>>(xcur, xcurb, NN * DD);
    for (int t = 0; t < 2; t++) {
      gemm(xcurb, WqT[t], tfp(t, 1), nullptr, nullptr, qb, NN, DD, DD, 0);
      gemm(xcurb, WkT[t], tfp(t, 3), nullptr, nullptr, kb, NN, DD, DD, 0);
      gemm(xcurb, WvT[t], tfp(t, 5), nullptr, vf, nullptr, NN, DD, DD, 0);
      k_act_t<<<CDIV(NN * DD, 256), 256, 0, stream>>>(vf, vtb, NN, DD);
      if (biasp)
        k_attn<1><<<dim3(NN / 16, HH), 32, 0, stream>>>(qb, kb, vtb, biasp,
                                                        aof, NN);
      else
        k_attn<0><<<dim3(NN / 16, HH), 32, 0, stream>>>(qb, kb, vtb, nullptr,
                                                        aof, NN);
      k_f2b<<<CDIV(NN * DD, 256), 256, 0, stream>>>(aof, aob, NN * DD);
      gemm(aob, WoT[t], tfp(t, 7), nullptr, obuf, nullptr, NN, DD, DD, 0);
      ln(obuf, xcur, nullptr, tfp(t, 8), tfp(t, 9), x2, x2b);
      gemm(x2b, W1T[t], tfp(t, 13), nullptr, nullptr, midb, NN, 1024, DD, 1);
      gemm(midb, W2T[t], tfp(t, 15), nullptr, fbuf, nullptr, NN, DD, 1024, 0);
      ln(fbuf, x2, nullptr, tfp(t, 10), tfp(t, 11), xcur, xcurb);
    }
  };

  // pass 1 (no bias) -> emb in xcur / xcurb
  encoder(nullptr);

  // ---- MLP head -> coords0 ----
  gemm(xcurb, mW1T, mlpb1, nullptr, nullptr, t1b, NN, 512, DD, 1);
  gemm(t1b, mW2T, mlpb2, nullptr, t2f, nullptr, NN, DD, 512, 1);
  k_proj3<<<CDIV(NN * 3, 256), 256, 0, stream>>>(t2f, mlpW3, mlpb3, nullptr, c0,
                                                 DD);

  // ---- refinement iteration ----
  k_struct_bias<<<CDIV(NN * NN, 256), 256, 0, stream>>>(c0, biasb);
  encoder(biasb);
  k_concat_pad<<<CDIV(NN * 288, 256), 256, 0, stream>>>(xcur, c0, hcatb);
  gemm(hcatb, rW1T, refb1, nullptr, d1f, nullptr, NN, DD, 288, 1);
  k_proj3<<<CDIV(NN * 3, 256), 256, 0, stream>>>(d1f, refW2, refb2, c0, c1, DD);

  // ---- outputs: coords (final) + stacked all_coords [c0, c1] ----
  float* out = (float*)d_out;
  k_copy<<<CDIV(NN * 3, 256), 256, 0, stream>>>(c1, out, NN * 3);
  k_copy<<<CDIV(NN * 3, 256), 256, 0, stream>>>(c0, out + NN * 3, NN * 3);
  k_copy<<<CDIV(NN * 3, 256), 256, 0, stream>>>(c1, out + 2 * NN * 3, NN * 3);
}
